// FavorAttentionLayer_83382495085172
// MI455X (gfx1250) — matrix operand order repaired and checked by reading, not compiled, not run on hardware
//
#include <hip/hip_runtime.h>
#include <hip/hip_bf16.h>
#include <math.h>

// Problem constants (match reference)
#define BB 4
#define SS 2048
#define EE 1024
#define HH 16
#define DD 64
#define MM 256
#define BS (BB*SS)   // 8192 rows
#define BH (BB*HH)   // 64 (b,h) pairs
#define CT 32        // scan chunk (timesteps) -> K=32 WMMA updates
#define QS 264       // padded LDS row stride for 256-wide bf16 tiles (528B)
#define TS 40        // padded LDS row stride for 32-wide bf16 tiles (80B)

typedef __bf16 bf16_t;
typedef __attribute__((ext_vector_type(16))) __bf16 v16bf;
typedef __attribute__((ext_vector_type(8)))  __bf16 v8bf;
typedef __attribute__((ext_vector_type(8)))  float  v8f;

// Fragment gathers per CDNA5 ISA 7.12.2 (16-bit A 16x32 / B 32x16 lane tables).
// A lanes<16: K {0..7,16..23}; lanes>=16: K {8..15,24..31}
__device__ __forceinline__ v16bf frag_a(const bf16_t* row, int k0, int half) {
  v8bf a0 = *(const v8bf*)(row + k0 + half * 8);
  v8bf a1 = *(const v8bf*)(row + k0 + 16 + half * 8);
  v16bf r;
#pragma unroll
  for (int i = 0; i < 8; ++i) { r[i] = a0[i]; r[8 + i] = a1[i]; }
  return r;
}
// B lanes<16: K {0..7,16..23}; lanes>=16: K {8..15,24..31}  (row = column of the B matrix)
__device__ __forceinline__ v16bf frag_b(const bf16_t* row, int k0, int half) {
  v8bf b0 = *(const v8bf*)(row + k0 + half * 8);
  v8bf b1 = *(const v8bf*)(row + k0 + 16 + half * 8);
  v16bf r;
#pragma unroll
  for (int i = 0; i < 8; ++i) { r[i] = b0[i]; r[8 + i] = b1[i]; }
  return r;
}
#define WMMA_BF16(A, B, C) \
  __builtin_amdgcn_wmma_f32_16x16x32_bf16(false, (A), false, (B), (short)0, (C), false, false)

// ---------------------------------------------------------------- conversions
__global__ void cvt_f32_to_bf16(const float* __restrict__ src,
                                bf16_t* __restrict__ dst, int n) {
  int i = blockIdx.x * blockDim.x + threadIdx.x;
  if (i < n) dst[i] = (bf16_t)src[i];
}

// Wt[n][k] = W[k][n]  (B-operand wants columns of W contiguous per lane)
__global__ void transpose_to_bf16(const float* __restrict__ W,
                                  bf16_t* __restrict__ Wt) {
  int i = blockIdx.x * blockDim.x + threadIdx.x;   // over EE*EE
  int n = i / EE, k = i % EE;
  Wt[(size_t)n * EE + k] = (bf16_t)W[(size_t)k * EE + n];
}

// ------------------------------------------------------------------ WMMA GEMM
// 2x2 register tiling: each wave computes 32x32; block (8 waves) = 128x64.
#define GM_QK  0   // outb <- bf16( (acc+bias) * 1/sqrt(D) ), permuted [B,H,S,D]
#define GM_V   1   // outf <- acc+bias, permuted [B,H,S,D]
#define GM_RES 2   // outf <- acc+bias+resid, plain [BS,EE]

__global__ __launch_bounds__(256)
void wmma_gemm_e(const bf16_t* __restrict__ A,    // [BS, EE]
                 const bf16_t* __restrict__ Wt,   // [EE, EE]
                 const float*  __restrict__ bias, // [EE]
                 const float*  __restrict__ resid,// [BS,EE] or null
                 float*  __restrict__ outf,
                 bf16_t* __restrict__ outb,
                 int mode)
{
  const int wave = threadIdx.x >> 5;
  const int lane = threadIdx.x & 31;
  const int half = lane >> 4;
  const int l    = lane & 15;
  const int wr   = wave >> 1, wc = wave & 1;
  const int r0   = blockIdx.y * 128 + wr * 32;
  const int n0   = blockIdx.x * 64  + wc * 32;

  const bf16_t* ar0 = A  + (size_t)(r0 + l) * EE;
  const bf16_t* ar1 = A  + (size_t)(r0 + 16 + l) * EE;
  const bf16_t* bc0 = Wt + (size_t)(n0 + l) * EE;
  const bf16_t* bc1 = Wt + (size_t)(n0 + 16 + l) * EE;

  v8f acc[2][2];
#pragma unroll
  for (int i = 0; i < 2; ++i)
#pragma unroll
    for (int j = 0; j < 2; ++j) acc[i][j] = v8f{0.f,0.f,0.f,0.f,0.f,0.f,0.f,0.f};

  for (int k0 = 0; k0 < EE; k0 += 32) {
    v16bf a0 = frag_a(ar0, k0, half);
    v16bf a1 = frag_a(ar1, k0, half);
    v16bf b0 = frag_b(bc0, k0, half);
    v16bf b1 = frag_b(bc1, k0, half);
    acc[0][0] = WMMA_BF16(a0, b0, acc[0][0]);
    acc[0][1] = WMMA_BF16(a0, b1, acc[0][1]);
    acc[1][0] = WMMA_BF16(a1, b0, acc[1][0]);
    acc[1][1] = WMMA_BF16(a1, b1, acc[1][1]);
  }
#pragma unroll
  for (int rr = 0; rr < 2; ++rr)
#pragma unroll
    for (int cc = 0; cc < 2; ++cc)
#pragma unroll
      for (int i = 0; i < 8; ++i) {
        int row = r0 + rr * 16 + i + half * 8;
        int col = n0 + cc * 16 + l;
        float v = acc[rr][cc][i] + bias[col];
        if (mode == GM_RES) {
          outf[(size_t)row * EE + col] = v + resid[(size_t)row * EE + col];
        } else {
          int b = row >> 11, s = row & (SS - 1);
          int h = col >> 6,  d = col & (DD - 1);
          size_t idx = (((size_t)(b * HH + h)) * SS + s) * DD + d;
          if (mode == GM_V) outf[idx] = v;
          else              outb[idx] = (bf16_t)(v * 0.125f);  // 1/sqrt(D)
        }
      }
}

// corr[b,h,s] = 0.5 * sum_d (scaled x)^2
__global__ void corr_kernel(const bf16_t* __restrict__ Xs,
                            float* __restrict__ corr) {
  int i = blockIdx.x * blockDim.x + threadIdx.x;   // over BH*SS
  const bf16_t* p = Xs + (size_t)i * DD;
  float s = 0.f;
#pragma unroll
  for (int d = 0; d < DD; ++d) { float x = (float)p[d]; s += x * x; }
  corr[i] = 0.5f * s;
}

// phi = (1/sqrt(M)) * exp( Xs @ proj^T - corr ), bf16 output (K=64, N=256)
__global__ __launch_bounds__(256)
void phi_wmma(const bf16_t* __restrict__ Xs,   // [BH,SS,DD] scaled bf16
              const bf16_t* __restrict__ Pb,   // [MM,DD] (already B-layout)
              const float*  __restrict__ corr, // [BH*SS]
              bf16_t* __restrict__ out)        // [BH,SS,MM] bf16
{
  const int wave = threadIdx.x >> 5;
  const int lane = threadIdx.x & 31;
  const int half = lane >> 4;
  const int l    = lane & 15;
  const int bh   = blockIdx.z;
  const int n0   = blockIdx.x * 16;
  const int r0   = (blockIdx.y * 8 + wave) * 16;

  const bf16_t* arow = Xs + ((size_t)bh * SS + r0 + l) * DD;
  const bf16_t* bcol = Pb + (size_t)(n0 + l) * DD;

  v8f acc = {0.f,0.f,0.f,0.f,0.f,0.f,0.f,0.f};
  for (int k0 = 0; k0 < DD; k0 += 32) {
    v16bf av = frag_a(arow, k0, half);
    v16bf bv = frag_b(bcol, k0, half);
    acc = WMMA_BF16(av, bv, acc);
  }
#pragma unroll
  for (int i = 0; i < 8; ++i) {
    int row = r0 + i + half * 8;
    int col = n0 + l;
    float v = 0.0625f * __expf(acc[i] - corr[(size_t)bh * SS + row]);
    out[((size_t)bh * SS + row) * MM + col] = (bf16_t)v;
  }
}

// ------------------------------------------------- chunked causal scan (WMMA)
// One WG per (b,h); 8 waves. kv state [D=64 x M=256] (transposed) lives in
// WMMA accumulators: wave w owns tiles (dt = w&3, mt = (w>>2)*8 + 0..7).
// Per 32-step chunk:
//   S    = tril(Qp @ Kp^T)                [32x32]  K=256
//   den  = rowsum(S) + Qp @ ksum_prev
//   num  = S @ V + Qp @ kv_prev^T         [32x64]  K=32 + K=256
//   kv_T += V^T @ Kp                      [64x256] K=32 (stays in registers)
// attn is written directly in merged [B,S,E] bf16 layout.
__global__ __launch_bounds__(256)
void favor_scan_wmma(const bf16_t* __restrict__ qp,  // [BH,SS,MM]
                     const bf16_t* __restrict__ kp,  // [BH,SS,MM]
                     const float*  __restrict__ V,   // [BH,SS,DD] f32
                     bf16_t* __restrict__ attnb)     // [BS,EE] bf16 (merged)
{
  extern __shared__ char smem_raw[];
  bf16_t* Qc  = (bf16_t*)smem_raw;       // [32][QS]
  bf16_t* Kc  = Qc  + 32 * QS;           // [32][QS]
  bf16_t* kvb = Kc  + 32 * QS;           // [64][QS]  bf16 copy of kv_T
  bf16_t* KT  = kvb + 64 * QS;           // [256][TS] Kp^T
  bf16_t* VT  = KT  + 256 * TS;          // [64][TS]  V^T
  bf16_t* Ab  = VT  + 64 * TS;           // [32][TS]  masked scores
  float*  ksum = (float*)(Ab + 32 * TS); // [256]
  float*  qdot = ksum + 256;             // [32]
  float*  denf = qdot + 32;              // [32]

  const int bh   = blockIdx.x;
  const int tid  = threadIdx.x;
  const int wave = tid >> 5;
  const int lane = tid & 31;
  const int half = lane >> 4;
  const int l    = lane & 15;
  const int dt   = wave & 3;      // kv d-tile (G3)
  const int mh   = wave >> 2;     // kv m-tile half (G3)
  const int b_   = bh >> 4, h_ = bh & 15;

  const bf16_t* qpb = qp + (size_t)bh * SS * MM;
  const bf16_t* kpb = kp + (size_t)bh * SS * MM;
  const float*  vb  = V  + (size_t)bh * SS * DD;

  v8f kvacc[8];
#pragma unroll
  for (int j = 0; j < 8; ++j) kvacc[j] = v8f{0.f,0.f,0.f,0.f,0.f,0.f,0.f,0.f};
  ksum[tid] = 0.f;

  for (int c = 0; c < SS / CT; ++c) {
    // ---- phase 0: stage chunk tiles + dump kv accumulators to bf16 LDS ----
    const size_t qbase = (size_t)c * CT * MM;
    for (int u = tid; u < CT * 32; u += 256) {        // 32 rows x 32 v8bf units
      int t = u >> 5, seg = u & 31;
      *(v8bf*)(Qc + t * QS + seg * 8) = *(const v8bf*)(qpb + qbase + t * MM + seg * 8);
      *(v8bf*)(Kc + t * QS + seg * 8) = *(const v8bf*)(kpb + qbase + t * MM + seg * 8);
    }
    for (int u = tid; u < CT * MM; u += 256) {        // Kp^T
      int t = u >> 8, m = u & 255;
      KT[m * TS + t] = kpb[qbase + t * MM + m];
    }
    for (int u = tid; u < CT * DD; u += 256) {        // V^T (f32 -> bf16)
      int t = u >> 6, d = u & 63;
      VT[d * TS + t] = (bf16_t)vb[(size_t)c * CT * DD + t * DD + d];
    }
#pragma unroll
    for (int j = 0; j < 8; ++j)                       // kv_T bf16 snapshot
#pragma unroll
      for (int i = 0; i < 8; ++i)
        kvb[(dt * 16 + i + half * 8) * QS + (mh * 8 + j) * 16 + l] = (bf16_t)kvacc[j][i];
    __syncthreads();

    // ---- phase A: scores (waves 0-3), qdot = Qp @ ksum_prev (wave 4) ----
    if (wave < 4) {
      const int ti = wave >> 1, tj = wave & 1;
      if (!(ti == 0 && tj == 1)) {
        v8f s = {0.f,0.f,0.f,0.f,0.f,0.f,0.f,0.f};
        const bf16_t* qrow = Qc + (ti * 16 + l) * QS;
        const bf16_t* krow = Kc + (tj * 16 + l) * QS;
        for (int k0 = 0; k0 < MM; k0 += 32)
          s = WMMA_BF16(frag_a(qrow, k0, half), frag_b(krow, k0, half), s);
#pragma unroll
        for (int i = 0; i < 8; ++i) {                 // causal mask (inclusive)
          int rl = ti * 16 + i + half * 8, cl = tj * 16 + l;
          Ab[rl * TS + cl] = (bf16_t)((rl >= cl) ? s[i] : 0.f);
        }
      } else {                                        // strictly-upper tile = 0
#pragma unroll
        for (int i = 0; i < 8; ++i)
          Ab[(i + half * 8) * TS + 16 + l] = (bf16_t)0.f;
      }
    } else if (wave == 4) {
      int s = tid - 128;                              // 32 rows
      float a = 0.f;
      const bf16_t* qrow = Qc + s * QS;
      for (int m = 0; m < MM; ++m) a += (float)qrow[m] * ksum[m];
      qdot[s] = a;
    }
    __syncthreads();

    // ---- phase B: num = S@V + Qp@kv_prev; 8 waves, one 16x16 tile each ----
    const int rt = wave >> 2, d2 = wave & 3;
    v8f n = {0.f,0.f,0.f,0.f,0.f,0.f,0.f,0.f};
    {
      const bf16_t* arow = Ab + (rt * 16 + l) * TS;   // S tile rows, K=32
      const bf16_t* brow = VT + (d2 * 16 + l) * TS;
      n = WMMA_BF16(frag_a(arow, 0, half), frag_b(brow, 0, half), n);
      const bf16_t* qrow = Qc  + (rt * 16 + l) * QS;  // Qp @ kv_prev, K=256
      const bf16_t* kvr  = kvb + (d2 * 16 + l) * QS;
      for (int k0 = 0; k0 < MM; k0 += 32)
        n = WMMA_BF16(frag_a(qrow, k0, half), frag_b(kvr, k0, half), n);
    }
    if (tid < 32) {                                   // den = rowsum(S) + qdot
      float rs = 0.f;
#pragma unroll
      for (int t = 0; t < CT; ++t) rs += (float)Ab[tid * TS + t];
      denf[tid] = rs + qdot[tid];
    }
    __syncthreads();
#pragma unroll
    for (int i = 0; i < 8; ++i) {                     // write merged bf16 attn
      int sl = rt * 16 + i + half * 8;
      size_t row = (size_t)b_ * SS + c * CT + sl;
      size_t col = (size_t)h_ * DD + d2 * 16 + l;
      attnb[row * EE + col] = (bf16_t)(n[i] / denf[sl]);
    }

    // ---- phase C: kv_T += V^T @ Kp (registers); ksum += colsum(Kp) ----
    {
      float s2 = 0.f;
#pragma unroll
      for (int t = 0; t < CT; ++t) s2 += (float)KT[tid * TS + t];
      ksum[tid] += s2;
    }
    {
      const bf16_t* arow = VT + (dt * 16 + l) * TS;
      v16bf av = frag_a(arow, 0, half);
#pragma unroll
      for (int j = 0; j < 8; ++j) {
        const bf16_t* brow = KT + ((mh * 8 + j) * 16 + l) * TS;
        kvacc[j] = WMMA_BF16(av, frag_b(brow, 0, half), kvacc[j]);
      }
    }
    __syncthreads();
  }
}

// --------------------------------------------------------------- layernorm
__global__ __launch_bounds__(256)
void layernorm(const float* __restrict__ res, const float* __restrict__ gamma,
               const float* __restrict__ beta, float* __restrict__ out) {
  __shared__ float red[256];
  const int r = blockIdx.x;
  const float* row = res + (size_t)r * EE;
  float lv[4]; float s = 0.f;
#pragma unroll
  for (int i = 0; i < 4; ++i) { lv[i] = row[threadIdx.x + i * 256]; s += lv[i]; }
  red[threadIdx.x] = s; __syncthreads();
  for (int off = 128; off > 0; off >>= 1) {
    if (threadIdx.x < off) red[threadIdx.x] += red[threadIdx.x + off];
    __syncthreads();
  }
  float mu = red[0] * (1.f / EE);
  __syncthreads();
  float vs = 0.f;
#pragma unroll
  for (int i = 0; i < 4; ++i) { float d = lv[i] - mu; vs += d * d; }
  red[threadIdx.x] = vs; __syncthreads();
  for (int off = 128; off > 0; off >>= 1) {
    if (threadIdx.x < off) red[threadIdx.x] += red[threadIdx.x + off];
    __syncthreads();
  }
  float inv = rsqrtf(red[0] * (1.f / EE) + 1e-6f);
#pragma unroll
  for (int i = 0; i < 4; ++i) {
    int c = threadIdx.x + i * 256;
    out[(size_t)r * EE + c] = gamma[c] * (lv[i] - mu) * inv + beta[c];
  }
}

// ------------------------------------------------------------------- driver
extern "C" void kernel_launch(void* const* d_in, const int* in_sizes, int n_in,
                              void* d_out, int out_size, void* d_ws, size_t ws_size,
                              hipStream_t stream) {
  (void)in_sizes; (void)n_in; (void)out_size; (void)ws_size;
  const float* x     = (const float*)d_in[0];
  const float* q_w   = (const float*)d_in[1];
  const float* q_b   = (const float*)d_in[2];
  const float* k_w   = (const float*)d_in[3];
  const float* k_b   = (const float*)d_in[4];
  const float* v_w   = (const float*)d_in[5];
  const float* v_b   = (const float*)d_in[6];
  const float* o_w   = (const float*)d_in[7];
  const float* o_b   = (const float*)d_in[8];
  const float* proj  = (const float*)d_in[9];
  const float* gamma = (const float*)d_in[10];
  const float* beta  = (const float*)d_in[11];
  float* out = (float*)d_out;

  char* ws = (char*)d_ws;
  size_t off = 0;
  auto alloc = [&](size_t bytes) -> void* {
    void* p = ws + off; off += (bytes + 255) & ~(size_t)255; return p;
  };
  bf16_t* xb    = (bf16_t*)alloc((size_t)BS * EE * 2);
  bf16_t* wtq   = (bf16_t*)alloc((size_t)EE * EE * 2);
  bf16_t* wtk   = (bf16_t*)alloc((size_t)EE * EE * 2);
  bf16_t* wtv   = (bf16_t*)alloc((size_t)EE * EE * 2);
  bf16_t* wto   = (bf16_t*)alloc((size_t)EE * EE * 2);
  bf16_t* projb = (bf16_t*)alloc((size_t)MM * DD * 2);
  float*  Vf    = (float*) alloc((size_t)BH * SS * DD * 4);
  bf16_t* Qsb   = (bf16_t*)alloc((size_t)BH * SS * DD * 2);
  bf16_t* Ksb   = (bf16_t*)alloc((size_t)BH * SS * DD * 2);
  float*  corrQ = (float*) alloc((size_t)BH * SS * 4);
  float*  corrK = (float*) alloc((size_t)BH * SS * 4);
  bf16_t* qpbuf = (bf16_t*)alloc((size_t)BH * SS * MM * 2);
  bf16_t* kpbuf = (bf16_t*)alloc((size_t)BH * SS * MM * 2);
  bf16_t* attnb = (bf16_t*)alloc((size_t)BS * EE * 2);
  float*  resf  = (float*) alloc((size_t)BS * EE * 4);

  cvt_f32_to_bf16<<<BS * EE / 256, 256, 0, stream>>>(x, xb, BS * EE);
  transpose_to_bf16<<<EE * EE / 256, 256, 0, stream>>>(q_w, wtq);
  transpose_to_bf16<<<EE * EE / 256, 256, 0, stream>>>(k_w, wtk);
  transpose_to_bf16<<<EE * EE / 256, 256, 0, stream>>>(v_w, wtv);
  transpose_to_bf16<<<EE * EE / 256, 256, 0, stream>>>(o_w, wto);
  cvt_f32_to_bf16<<<MM * DD / 256, 256, 0, stream>>>(proj, projb, MM * DD);

  dim3 gg(EE / 64, BS / 128);
  wmma_gemm_e<<<gg, 256, 0, stream>>>(xb, wtq, q_b, nullptr, nullptr, Qsb, GM_QK);
  wmma_gemm_e<<<gg, 256, 0, stream>>>(xb, wtk, k_b, nullptr, nullptr, Ksb, GM_QK);
  wmma_gemm_e<<<gg, 256, 0, stream>>>(xb, wtv, v_b, nullptr, Vf, nullptr, GM_V);

  corr_kernel<<<BH * SS / 256, 256, 0, stream>>>(Qsb, corrQ);
  corr_kernel<<<BH * SS / 256, 256, 0, stream>>>(Ksb, corrK);

  dim3 gp(MM / 16, SS / 128, BH);
  phi_wmma<<<gp, 256, 0, stream>>>(Qsb, projb, corrQ, qpbuf);
  phi_wmma<<<gp, 256, 0, stream>>>(Ksb, projb, corrK, kpbuf);

  size_t scan_smem = (size_t)(32 * QS + 32 * QS + 64 * QS + 256 * TS + 64 * TS + 32 * TS) * 2
                   + (size_t)(256 + 32 + 32) * 4;
  hipFuncSetAttribute((const void*)favor_scan_wmma,
                      hipFuncAttributeMaxDynamicSharedMemorySize, (int)scan_smem);
  favor_scan_wmma<<<BH, 256, scan_smem, stream>>>(qpbuf, kpbuf, Vf, attnb);

  wmma_gemm_e<<<gg, 256, 0, stream>>>(attnb, wto, o_b, x, resf, nullptr, GM_RES);
  layernorm<<<BS, 256, 0, stream>>>(resf, gamma, beta, out);
}
